// MambaSSM_65798898975185
// MI455X (gfx1250) — matrix operand order repaired and checked by reading, not compiled, not run on hardware
//
#include <hip/hip_runtime.h>
#include <hip/hip_bf16.h>

typedef __attribute__((ext_vector_type(16))) _Float16 v16h;
typedef __attribute__((ext_vector_type(8)))  _Float16 v8h;
typedef __attribute__((ext_vector_type(8)))  float    v8f;
typedef int v4i __attribute__((vector_size(16)));

#define L_SEQ   4096
#define D_MODEL 511
#define D_INNER 1022
#define DIP     1024      // padded d_inner
#define NXZ     2048      // padded 2*d_inner
#define KDM     512       // padded d_model (K of GEMM1)
#define DTR     32
#define DST     16
#define NCHUNK  64
#define CLEN    64        // NCHUNK*CLEN == L_SEQ
#define LDSS    40        // LDS f16 row stride (80B, 16B aligned, conflict-friendly)
#define BM      128
#define BN      64

// gfx1250 async global->LDS path (ASYNCcnt-tracked, no VGPR staging)
#if defined(__AMDGCN__) && __has_builtin(__builtin_amdgcn_global_load_async_to_lds_b128) && __has_builtin(__builtin_amdgcn_s_wait_asynccnt)
#define USE_ASYNC 1
#define GLD_ASYNC_B128(dst_lds, src_glb)                                        \
    __builtin_amdgcn_global_load_async_to_lds_b128(                             \
        (__attribute__((address_space(1))) v4i*)(src_glb),                      \
        (__attribute__((address_space(3))) v4i*)(dst_lds), 0, 0)
#else
#define USE_ASYNC 0
#endif

__device__ __forceinline__ float softplus_f(float x) {
    return (x > 20.0f) ? x : log1pf(expf(x));
}

// ---------------------------------------------------------------------------
// Pipelined f16 WMMA GEMM: C(MxN,f32) = A(MxK,f16 row-major) * Bt(NxK,f16)^T
// K mult of 32, M mult of 128, N mult of 64. Double-buffered LDS, one barrier
// per K-step, 4 wmma per wave per K-step. Tile fetch via async-to-LDS when
// available, else VGPR staging issued after the barrier.
// mode 0: plain store      mode 1: transposed store, cols < nreal
// mode 2: softplus(acc + bias[col]), zero for col >= nreal
// ---------------------------------------------------------------------------
__global__ void gemm_wmma_f16(const _Float16* __restrict__ A,
                              const _Float16* __restrict__ Bt,
                              float* __restrict__ C,
                              int K, int ldc, int mode, int nreal, int ldt,
                              const float* __restrict__ bias)
{
    __shared__ __align__(16) _Float16 As[2][BM * LDSS];
    __shared__ __align__(16) _Float16 Bs[2][BN * LDSS];

    const int tid  = threadIdx.x;
    const int wave = tid >> 5;
    const int lane = tid & 31;
    const int m16  = lane & 15;       // A row / B,C col within 16x16 tile
    const int g    = lane >> 4;       // half-wave group
    const int mt0  = (wave & 3) * 2;  // this wave's two m-tiles
    const int nt0  = (wave >> 2) * 2; // this wave's two n-tiles

    const int blockN = blockIdx.x * BN;
    const int blockM = blockIdx.y * BM;

    // staging: A 128x32 (16 f16/thread), B 64x32 (8 f16/thread)
    const int arow = tid >> 1;
    const int acol = (tid & 1) * 16;
    const int brow = tid >> 2;
    const int bcol = (tid & 3) * 8;

    const _Float16* Ag = A  + (size_t)(blockM + arow) * K + acol;
    const _Float16* Bg = Bt + (size_t)(blockN + brow) * K + bcol;

    v8f acc00 = {}, acc01 = {}, acc10 = {}, acc11 = {};

    const int nk = K >> 5;

#if USE_ASYNC
    // preamble: DMA tile 0 into buffer 0
    GLD_ASYNC_B128(&As[0][arow * LDSS + acol],     Ag);
    GLD_ASYNC_B128(&As[0][arow * LDSS + acol + 8], Ag + 8);
    GLD_ASYNC_B128(&Bs[0][brow * LDSS + bcol],     Bg);
#else
    {   // preamble: stage tile 0 into buffer 0
        v8h a0 = *(const v8h*)(Ag);
        v8h a1 = *(const v8h*)(Ag + 8);
        v8h b0 = *(const v8h*)(Bg);
        *(v8h*)(&As[0][arow * LDSS + acol])     = a0;
        *(v8h*)(&As[0][arow * LDSS + acol + 8]) = a1;
        *(v8h*)(&Bs[0][brow * LDSS + bcol])     = b0;
    }
    v8h av0, av1, bv;
#endif

    for (int k = 0; k < nk; ++k) {
        const int cur = k & 1;
        const int nxt = cur ^ 1;

#if USE_ASYNC
        __builtin_amdgcn_s_wait_asynccnt(0);   // my in-flight DMAs done
        __syncthreads();                       // => everyone's DMAs + reads done
        if (k + 1 < nk) {                      // DMA next tile during this tile's math
            const _Float16* Agn = Ag + ((k + 1) << 5);
            const _Float16* Bgn = Bg + ((k + 1) << 5);
            GLD_ASYNC_B128(&As[nxt][arow * LDSS + acol],     Agn);
            GLD_ASYNC_B128(&As[nxt][arow * LDSS + acol + 8], Agn + 8);
            GLD_ASYNC_B128(&Bs[nxt][brow * LDSS + bcol],     Bgn);
        }
#else
        __syncthreads();                       // buf[cur] ready; reads of buf[nxt] done
        if (k + 1 < nk) {                      // prefetch after barrier: latency hides under wmma
            const int k0 = (k + 1) << 5;
            av0 = *(const v8h*)(Ag + k0);
            av1 = *(const v8h*)(Ag + k0 + 8);
            bv  = *(const v8h*)(Bg + k0);
        }
#endif

        // A fragment: lane m16 = row; K = {8g..8g+7} U {16+8g..16+8g+7}
        const _Float16* ap0 = &As[cur][(mt0 * 16 + m16) * LDSS];
        const _Float16* ap1 = ap0 + 16 * LDSS;
        v8h a0lo = *(const v8h*)(ap0 + 8 * g);
        v8h a0hi = *(const v8h*)(ap0 + 16 + 8 * g);
        v8h a1lo = *(const v8h*)(ap1 + 8 * g);
        v8h a1hi = *(const v8h*)(ap1 + 16 + 8 * g);
        v16h a0 = __builtin_shufflevector(a0lo, a0hi, 0,1,2,3,4,5,6,7,8,9,10,11,12,13,14,15);
        v16h a1 = __builtin_shufflevector(a1lo, a1hi, 0,1,2,3,4,5,6,7,8,9,10,11,12,13,14,15);
        // B fragment: lane m16 = col; K = {8g..8g+7} U {16+8g..16+8g+7}
        const _Float16* bp0 = &Bs[cur][(nt0 * 16 + m16) * LDSS] + 8 * g;
        const _Float16* bp1 = bp0 + 16 * LDSS;
        v8h b0lo = *(const v8h*)(bp0);
        v8h b0hi = *(const v8h*)(bp0 + 16);
        v8h b1lo = *(const v8h*)(bp1);
        v8h b1hi = *(const v8h*)(bp1 + 16);
        v16h b0 = __builtin_shufflevector(b0lo, b0hi, 0,1,2,3,4,5,6,7,8,9,10,11,12,13,14,15);
        v16h b1 = __builtin_shufflevector(b1lo, b1hi, 0,1,2,3,4,5,6,7,8,9,10,11,12,13,14,15);

        acc00 = __builtin_amdgcn_wmma_f32_16x16x32_f16(false, a0, false, b0, (short)0, acc00, false, false);
        acc01 = __builtin_amdgcn_wmma_f32_16x16x32_f16(false, a0, false, b1, (short)0, acc01, false, false);
        acc10 = __builtin_amdgcn_wmma_f32_16x16x32_f16(false, a1, false, b0, (short)0, acc10, false, false);
        acc11 = __builtin_amdgcn_wmma_f32_16x16x32_f16(false, a1, false, b1, (short)0, acc11, false, false);

#if !USE_ASYNC
        if (k + 1 < nk) {                 // stage next tile into the other buffer
            *(v8h*)(&As[nxt][arow * LDSS + acol])     = av0;
            *(v8h*)(&As[nxt][arow * LDSS + acol + 8]) = av1;
            *(v8h*)(&Bs[nxt][brow * LDSS + bcol])     = bv;
        }
#endif
    }

    // C layout: element r of v8f -> M = r + 8*g, N = m16
    const int mb0 = blockM + mt0 * 16 + 8 * g;
    const int mb1 = mb0 + 16;
    const int c0  = blockN + nt0 * 16 + m16;
    const int c1  = c0 + 16;
    if (mode == 0) {
        #pragma unroll
        for (int r = 0; r < 8; ++r) {
            C[(size_t)(mb0 + r) * ldc + c0] = acc00[r];
            C[(size_t)(mb0 + r) * ldc + c1] = acc01[r];
            C[(size_t)(mb1 + r) * ldc + c0] = acc10[r];
            C[(size_t)(mb1 + r) * ldc + c1] = acc11[r];
        }
    } else if (mode == 1) {
        #pragma unroll
        for (int r = 0; r < 8; ++r) {
            if (c0 < nreal) {
                C[(size_t)c0 * ldt + (mb0 + r)] = acc00[r];
                C[(size_t)c0 * ldt + (mb1 + r)] = acc10[r];
            }
            if (c1 < nreal) {
                C[(size_t)c1 * ldt + (mb0 + r)] = acc01[r];
                C[(size_t)c1 * ldt + (mb1 + r)] = acc11[r];
            }
        }
    } else {
        const bool v0 = (c0 < nreal), v1 = (c1 < nreal);
        const float bi0 = v0 ? bias[c0] : 0.0f;
        const float bi1 = v1 ? bias[c1] : 0.0f;
        #pragma unroll
        for (int r = 0; r < 8; ++r) {
            C[(size_t)(mb0 + r) * ldc + c0] = v0 ? softplus_f(acc00[r] + bi0) : 0.0f;
            C[(size_t)(mb0 + r) * ldc + c1] = v1 ? softplus_f(acc01[r] + bi1) : 0.0f;
            C[(size_t)(mb1 + r) * ldc + c0] = v0 ? softplus_f(acc10[r] + bi0) : 0.0f;
            C[(size_t)(mb1 + r) * ldc + c1] = v1 ? softplus_f(acc11[r] + bi1) : 0.0f;
        }
    }
}

// ---------------------------------------------------------------------------
// Transpose + convert + zero-pad: dst[r*Kp + k] = (k<Kreal && r<Rreal) ? f16(src[k*lds + r]) : 0
// ---------------------------------------------------------------------------
__global__ void convert_transpose(const float* __restrict__ src, _Float16* __restrict__ dst,
                                  int lds, int Kreal, int Rreal, int Kp)
{
    int k = blockIdx.x * 64 + threadIdx.x;
    int r = blockIdx.y * 4 + threadIdx.y;
    if (k >= Kp) return;
    float v = (k < Kreal && r < Rreal) ? src[(size_t)k * lds + r] : 0.0f;
    dst[(size_t)r * Kp + k] = (_Float16)v;
}

// ---------------------------------------------------------------------------
// Causal depthwise conv (width 4) + SiLU. xz (L,NXZ) f32; x = cols 0..1021.
// ---------------------------------------------------------------------------
__global__ void conv_silu(const float* __restrict__ xz,
                          const float* __restrict__ conv_w,
                          const float* __restrict__ conv_b,
                          float* __restrict__ xc32, _Float16* __restrict__ xch)
{
    int idx = blockIdx.x * 256 + threadIdx.x;        // over L*DIP
    int l = idx >> 10;
    int c = idx & (DIP - 1);
    float v = 0.0f;
    if (c < D_INNER) {
        float acc = conv_b[c];
        #pragma unroll
        for (int k = 0; k < 4; ++k) {
            int lt = l + k - 3;
            if (lt >= 0) acc += xz[(size_t)lt * NXZ + c] * conv_w[c * 4 + k];
        }
        v = acc / (1.0f + expf(-acc));               // SiLU
    }
    xc32[idx] = v;
    xch[idx]  = (_Float16)v;
}

// dt slice of x_dbl -> f16 (L,32)
__global__ void dt_to_f16(const float* __restrict__ xdbl, _Float16* __restrict__ dth)
{
    int idx = blockIdx.x * 256 + threadIdx.x;        // over L*32
    int l = idx >> 5, j = idx & 31;
    dth[idx] = (_Float16)xdbl[(size_t)l * 64 + j];
}

// ---------------------------------------------------------------------------
// Chunked parallel scan.  Per (chunk, channel) thread owns 16 states.
// ---------------------------------------------------------------------------
__global__ void scan_phaseA(const float* __restrict__ delta,
                            const float* __restrict__ xc32,
                            const float* __restrict__ xdbl,
                            const float* __restrict__ A_log,
                            float* __restrict__ P, float* __restrict__ Hend)
{
    __shared__ float bc[CLEN * 32];                  // B(16)+C(16) per step
    const int ch = blockIdx.y;
    const int c  = blockIdx.x * 128 + threadIdx.x;   // channel (0..1023)
    const int t0 = ch * CLEN;
    for (int i = threadIdx.x; i < CLEN * 32; i += 128) {
        int t = i >> 5, j = i & 31;
        bc[i] = xdbl[(size_t)(t0 + t) * 64 + 32 + j];
    }
    __syncthreads();

    const bool valid = (c < D_INNER);
    float a2[DST], h[DST], Pp[DST];
    #pragma unroll
    for (int s = 0; s < DST; ++s) {
        a2[s] = valid ? (-expf(A_log[c * DST + s]) * 1.44269504f) : 0.0f;
        h[s] = 0.0f; Pp[s] = 1.0f;
    }
    for (int t = 0; t < CLEN; ++t) {
        int tg = t0 + t;
        float d  = delta[(size_t)tg * DIP + c];
        float xc = xc32[(size_t)tg * DIP + c];
        float dx = d * xc;
        #pragma unroll
        for (int s = 0; s < DST; ++s) {
            float dA = exp2f(d * a2[s]);
            h[s]  = dA * h[s] + dx * bc[t * 32 + s];
            Pp[s] *= dA;
        }
    }
    size_t base = ((size_t)ch * DIP + c) * DST;
    #pragma unroll
    for (int s = 0; s < DST; ++s) { P[base + s] = Pp[s]; Hend[base + s] = h[s]; }
}

__global__ void scan_phaseB(const float* __restrict__ P,
                            const float* __restrict__ Hend,
                            float* __restrict__ Hstart)
{
    int idx = blockIdx.x * 256 + threadIdx.x;        // (c,s): 0..16383
    float h = 0.0f;
    for (int ch = 0; ch < NCHUNK; ++ch) {
        size_t o = (size_t)ch * (DIP * DST) + idx;
        Hstart[o] = h;
        h = P[o] * h + Hend[o];
    }
}

__global__ void scan_phaseC(const float* __restrict__ delta,
                            const float* __restrict__ xc32,
                            const float* __restrict__ xdbl,
                            const float* __restrict__ xz,
                            const float* __restrict__ A_log,
                            const float* __restrict__ Dp,
                            const float* __restrict__ Hstart,
                            _Float16* __restrict__ yh)
{
    __shared__ float bc[CLEN * 32];
    const int ch = blockIdx.y;
    const int c  = blockIdx.x * 128 + threadIdx.x;
    const int t0 = ch * CLEN;
    for (int i = threadIdx.x; i < CLEN * 32; i += 128) {
        int t = i >> 5, j = i & 31;
        bc[i] = xdbl[(size_t)(t0 + t) * 64 + 32 + j];
    }
    __syncthreads();

    const bool valid = (c < D_INNER);
    const float dcoef = valid ? Dp[c] : 0.0f;
    float a2[DST], h[DST];
    size_t base = ((size_t)ch * DIP + c) * DST;
    #pragma unroll
    for (int s = 0; s < DST; ++s) {
        a2[s] = valid ? (-expf(A_log[c * DST + s]) * 1.44269504f) : 0.0f;
        h[s] = Hstart[base + s];
    }
    for (int t = 0; t < CLEN; ++t) {
        int tg = t0 + t;
        float d  = delta[(size_t)tg * DIP + c];
        float xc = xc32[(size_t)tg * DIP + c];
        float dx = d * xc;
        float y = 0.0f;
        #pragma unroll
        for (int s = 0; s < DST; ++s) {
            float dA = exp2f(d * a2[s]);
            h[s] = dA * h[s] + dx * bc[t * 32 + s];
            y += h[s] * bc[t * 32 + 16 + s];
        }
        float out = 0.0f;
        if (valid) {
            float z  = xz[(size_t)tg * NXZ + D_INNER + c];
            float sz = z / (1.0f + expf(-z));
            out = (y + xc * dcoef) * sz;
        }
        yh[(size_t)tg * DIP + c] = (_Float16)out;
    }
}

// ---------------------------------------------------------------------------
static inline size_t align256(size_t x) { return (x + 255) & ~(size_t)255; }

extern "C" void kernel_launch(void* const* d_in, const int* in_sizes, int n_in,
                              void* d_out, int out_size, void* d_ws, size_t ws_size,
                              hipStream_t stream)
{
    (void)in_sizes; (void)n_in; (void)out_size; (void)ws_size;

    const float* Z      = (const float*)d_in[0];   // (511, 4096)
    const float* W_in   = (const float*)d_in[1];   // (511, 2044)
    const float* conv_w = (const float*)d_in[2];   // (1022, 4)
    const float* conv_b = (const float*)d_in[3];   // (1022)
    const float* W_x    = (const float*)d_in[4];   // (1022, 64)
    const float* W_dt   = (const float*)d_in[5];   // (32, 1022)
    const float* b_dt   = (const float*)d_in[6];   // (1022)
    const float* A_log  = (const float*)d_in[7];   // (1022, 16)
    const float* Dp     = (const float*)d_in[8];   // (1022)
    const float* W_out  = (const float*)d_in[9];   // (1022, 511)
    float* out = (float*)d_out;                    // (511, 4096)

    char* w = (char*)d_ws;
    size_t off = 0;
    auto alloc = [&](size_t bytes) { size_t o = off; off = align256(off + bytes); return (void*)(w + o); };

    _Float16* Zh      = (_Float16*)alloc((size_t)L_SEQ * KDM * 2);   // (L,512)
    _Float16* Wint    = (_Float16*)alloc((size_t)NXZ * KDM * 2);     // (2048,512)
    float*    xz      = (float*)   alloc((size_t)L_SEQ * NXZ * 4);   // (L,2048)
    float*    xc32    = (float*)   alloc((size_t)L_SEQ * DIP * 4);   // (L,1024)
    _Float16* xch     = (_Float16*)alloc((size_t)L_SEQ * DIP * 2);
    _Float16* Wxt     = (_Float16*)alloc((size_t)64 * DIP * 2);      // (64,1024)
    float*    xdbl    = (float*)   alloc((size_t)L_SEQ * 64 * 4);    // (L,64)
    _Float16* dth     = (_Float16*)alloc((size_t)L_SEQ * DTR * 2);   // (L,32)
    _Float16* Wdtt    = (_Float16*)alloc((size_t)DIP * DTR * 2);     // (1024,32)
    float*    delta   = (float*)   alloc((size_t)L_SEQ * DIP * 4);   // (L,1024), softplus fused in GEMM3
    float*    Pm      = (float*)   alloc((size_t)NCHUNK * DIP * DST * 4);
    float*    Hend    = (float*)   alloc((size_t)NCHUNK * DIP * DST * 4);
    float*    Hstart  = (float*)   alloc((size_t)NCHUNK * DIP * DST * 4);
    _Float16* yh      = (_Float16*)alloc((size_t)L_SEQ * DIP * 2);
    _Float16* Woutt   = (_Float16*)alloc((size_t)KDM * DIP * 2);     // (512,1024)

    dim3 cvtBlk(64, 4);
    convert_transpose<<<dim3(KDM / 64, L_SEQ / 4), cvtBlk, 0, stream>>>(Z, Zh, L_SEQ, D_MODEL, L_SEQ, KDM);
    convert_transpose<<<dim3(KDM / 64, NXZ / 4), cvtBlk, 0, stream>>>(W_in, Wint, 2 * D_INNER, D_MODEL, 2 * D_INNER, KDM);
    convert_transpose<<<dim3(DIP / 64, 64 / 4), cvtBlk, 0, stream>>>(W_x, Wxt, 64, D_INNER, 64, DIP);
    convert_transpose<<<dim3(1, DIP / 4), cvtBlk, 0, stream>>>(W_dt, Wdtt, D_INNER, DTR, D_INNER, DTR);
    convert_transpose<<<dim3(DIP / 64, KDM / 4), cvtBlk, 0, stream>>>(W_out, Woutt, D_MODEL, D_INNER, D_MODEL, DIP);

    // GEMM1: xz(L,2048) = Zh(L,512) * Wint^T
    gemm_wmma_f16<<<dim3(NXZ / BN, L_SEQ / BM), 256, 0, stream>>>(Zh, Wint, xz, KDM, NXZ, 0, 0, 0, nullptr);

    // conv + SiLU
    conv_silu<<<(L_SEQ * DIP) / 256, 256, 0, stream>>>(xz, conv_w, conv_b, xc32, xch);

    // GEMM2: xdbl(L,64) = xch(L,1024) * Wxt^T
    gemm_wmma_f16<<<dim3(1, L_SEQ / BM), 256, 0, stream>>>(xch, Wxt, xdbl, DIP, 64, 0, 0, 0, nullptr);

    // dt slice -> f16
    dt_to_f16<<<(L_SEQ * DTR) / 256, 256, 0, stream>>>(xdbl, dth);

    // GEMM3: delta(L,1024) = softplus(dth(L,32) * Wdtt^T + b_dt), zero pad cols
    gemm_wmma_f16<<<dim3(DIP / BN, L_SEQ / BM), 256, 0, stream>>>(dth, Wdtt, delta, DTR, DIP, 2, D_INNER, 0, b_dt);

    // chunked scan
    scan_phaseA<<<dim3(DIP / 128, NCHUNK), 128, 0, stream>>>(delta, xc32, xdbl, A_log, Pm, Hend);
    scan_phaseB<<<(DIP * DST) / 256, 256, 0, stream>>>(Pm, Hend, Hstart);
    scan_phaseC<<<dim3(DIP / 128, NCHUNK), 128, 0, stream>>>(delta, xc32, xdbl, xz, A_log, Dp, Hstart, yh);

    // GEMM4: out(d,l) = (yh * Woutt^T)^T, d < 511
    gemm_wmma_f16<<<dim3(KDM / BN, L_SEQ / BM), 256, 0, stream>>>(yh, Woutt, out, DIP, KDM, 1, D_MODEL, L_SEQ, nullptr);
}
